// HyperConv_85529978732823
// MI455X (gfx1250) — matrix operand order repaired and checked by reading, not compiled, not run on hardware
//
#include <hip/hip_runtime.h>
#include <hip/hip_bf16.h>

// Problem constants
#define BDIM 4
#define ZDIM 64
#define KCH  400
#define TDIM 48000
#define TK   120          // TDIM / KCH
#define CINC 128
#define COUTC 128
#define KSZ  3
#define HD   32
#define CKS  (CINC*KSZ)          // 384
#define RDIM (CINC*COUTC*KSZ)    // 49152
#define NCHUNK (BDIM*KCH)        // 1600

typedef __attribute__((ext_vector_type(16))) __bf16 v16bf;
typedef __attribute__((ext_vector_type(8)))  __bf16 v8bf;
typedef __attribute__((ext_vector_type(8)))  float  v8f;
typedef __attribute__((ext_vector_type(4)))  float  v4f;

#define SHUF16(lo, hi) __builtin_shufflevector(lo, hi, 0,1,2,3,4,5,6,7,8,9,10,11,12,13,14,15)

// CDNA5 async bulk copy: global -> LDS without touching VGPRs (ASYNCcnt).
static __device__ __forceinline__ void async_load_b128(unsigned lds_byte_off,
                                                       const void* gaddr) {
  asm volatile("global_load_async_to_lds_b128 %0, %1, off"
               :: "v"(lds_byte_off), "v"((unsigned long long)(uintptr_t)gaddr)
               : "memory");
}
static __device__ __forceinline__ void async_wait0() {
#if __has_builtin(__builtin_amdgcn_s_wait_asynccnt)
  __builtin_amdgcn_s_wait_asynccnt(0);
#else
  asm volatile("s_wait_asynccnt 0x0" ::: "memory");
#endif
}

// ---------------------------------------------------------------------------
// Kernel 1: per-chunk hidden vector h (weight MLP layer 1) stored as
// Hmat[H][chunk] bf16 (== WMMA B-fragment layout for wgen), plus the full
// bias MLP (tiny). One block (64 threads) per chunk.
// ---------------------------------------------------------------------------
__global__ void __launch_bounds__(64) head_kernel(
    const float* __restrict__ z,
    const float* __restrict__ w1,  const float* __restrict__ b1,
    const float* __restrict__ bw1, const float* __restrict__ bb1,
    const float* __restrict__ bw2, const float* __restrict__ bb2,
    __bf16* __restrict__ Hmat, float* __restrict__ biasbuf) {
  const int chunk = blockIdx.x;
  const int b = chunk / KCH, k = chunk % KCH;
  const int tid = threadIdx.x;
  __shared__ float hb[HD];
  const float* zc = z + (size_t)b * ZDIM * KCH + k;   // z[b][zz][k], stride KCH
  if (tid < HD) {
    float acc = b1[tid];
    for (int zz = 0; zz < ZDIM; ++zz) acc += w1[tid * ZDIM + zz] * zc[zz * KCH];
    Hmat[tid * NCHUNK + chunk] = (__bf16)fmaxf(acc, 0.0f);
  } else {
    const int j = tid - HD;
    float acc = bb1[j];
    for (int zz = 0; zz < ZDIM; ++zz) acc += bw1[j * ZDIM + zz] * zc[zz * KCH];
    hb[j] = fmaxf(acc, 0.0f);
  }
  __syncthreads();
  for (int o = tid; o < COUTC; o += 64) {
    float acc = bb2[o];
    for (int j = 0; j < HD; ++j) acc += bw2[o * HD + j] * hb[j];
    biasbuf[(size_t)chunk * COUTC + o] = acc;
  }
}

// ---------------------------------------------------------------------------
// Kernel 2: Wall[chunk][r] (r = cin_ks*128 + cout, bf16) = w2 @ h + b2.
// One v_wmma_f32_16x16x32_bf16 per 16x16 output tile (K == H == 32).
// D tiles bounce through a 512B per-wave LDS stage so global writes are
// coalesced b128 in Wall's [chunk][r] layout (== conv B-matrix layout).
// ---------------------------------------------------------------------------
__global__ void __launch_bounds__(256) wgen_kernel(
    const float* __restrict__ w2, const float* __restrict__ b2,
    const __bf16* __restrict__ Hmat, __bf16* __restrict__ Wall) {
  const int cbase = blockIdx.x * 16;
  const int tid = threadIdx.x;
  const int wave = tid >> 5, lane = tid & 31;
  const int l16 = lane & 15, lhalf = lane >> 4;
  __shared__ __align__(16) __bf16 stage[8][256];   // per-wave 16x16 bf16 tile

  // B fragment (constant for the whole block): lane column = chunk cbase+l16,
  // K (h index) {0..7,16..23} / {8..15,24..31} along VGPRs.
  const __bf16* hp = Hmat + cbase + l16;
  v16bf bfrag;
#pragma unroll
  for (int i = 0; i < 8; ++i) {
    bfrag[i]     = hp[(lhalf * 8 + i) * NCHUNK];
    bfrag[8 + i] = hp[(16 + lhalf * 8 + i) * NCHUNK];
  }

  __bf16* st = stage[wave];
#pragma unroll 4
  for (int it = 0; it < 24; ++it) {
    const int rbase = blockIdx.y * 3072 + (it * 8 + wave) * 16;
    // A fragment: lane row = rbase+l16; K chunks {0..7,16..23} / {8..15,24..31}
    const float* ap = w2 + (size_t)(rbase + l16) * HD + lhalf * 8;
    v4f a0 = *(const v4f*)(ap);
    v4f a1 = *(const v4f*)(ap + 4);
    v4f a2 = *(const v4f*)(ap + 16);
    v4f a3 = *(const v4f*)(ap + 20);
    v16bf afrag;
#pragma unroll
    for (int i = 0; i < 4; ++i) {
      afrag[i]      = (__bf16)a0[i];
      afrag[4 + i]  = (__bf16)a1[i];
      afrag[8 + i]  = (__bf16)a2[i];
      afrag[12 + i] = (__bf16)a3[i];
    }
    v8f c = {};
    c = __builtin_amdgcn_wmma_f32_16x16x32_bf16(false, afrag, false, bfrag,
                                                (short)0, c, false, false);
    // Bias add (b2[rbase + i + 8*lhalf]) then stage as [chunk][r] bf16.
    v4f bv0 = *(const v4f*)(b2 + rbase + lhalf * 8);
    v4f bv1 = *(const v4f*)(b2 + rbase + lhalf * 8 + 4);
#pragma unroll
    for (int i = 0; i < 8; ++i) {
      float v = c[i] + (i < 4 ? bv0[i] : bv1[i - 4]);
      st[l16 * 16 + (i + 8 * lhalf)] = (__bf16)v;    // stage[chunkcol][r_in_tile]
    }
    // Coalesced writeback: lanes 0..15 each own one chunk row (32B).
    if (lane < 16) {
      const v8bf* sp = (const v8bf*)(st + lane * 16);
      v8bf r0 = sp[0], r1 = sp[1];
      __bf16* wp = Wall + (size_t)(cbase + lane) * RDIM + rbase;
      *(v8bf*)(wp)     = r0;
      *(v8bf*)(wp + 8) = r1;
    }
  }
}

// ---------------------------------------------------------------------------
// Kernel 3: main hyper-conv. One block (8 waves) per chunk (b,k).
//   y[120,128] = xr[120,384] @ wc[384,128] + bias
// x window staged in LDS bf16 [time 0..129][cin] (causal 2-row zero pad) —
// the tap shift moves the LDS *row*, so every ds_load_b128 stays 16B aligned.
// Weight slab (96KB) streamed straight into LDS with
// global_load_async_to_lds_b128 (ASYNCcnt), overlapping the x staging.
// Wave tiling 4M x 2N: per K-step 4 A-frag + 2 B-frag LDS loads feed 8 WMMAs.
// Output bounced through per-wave LDS stage -> non-temporal coalesced b128.
// ---------------------------------------------------------------------------
#define XS_BYTES 33280          // [130][128] bf16
#define WC_BYTES 98304          // [384][128] bf16
__global__ void __launch_bounds__(256) conv_kernel(
    const float* __restrict__ x, const __bf16* __restrict__ Wall,
    const float* __restrict__ biasbuf, float* __restrict__ y) {
  const int chunk = blockIdx.x;
  const int b = chunk / KCH, k = chunk % KCH;
  const int tid = threadIdx.x;
  const int wave = tid >> 5, lane = tid & 31;
  const int l16 = lane & 15, lhalf = lane >> 4;

  __shared__ __align__(16) char smem[XS_BYTES + WC_BYTES];
  __bf16* xs = (__bf16*)smem;                 // [130][128] bf16
  __bf16* wc = (__bf16*)(smem + XS_BYTES);    // [384][128] bf16

  // ---- kick off async weight-slab stream: global -> LDS, no VGPR round-trip
  {
    const __bf16* wp = Wall + (size_t)chunk * RDIM;
#pragma unroll 4
    for (int jj = 0; jj < 24; ++jj) {
      const int idx = (jj * 256 + tid) * 8;                // bf16 elements
      async_load_b128((unsigned)(XS_BYTES + idx * 2), wp + idx);
    }
  }
  // ---- meanwhile stage x window with f32->bf16 conversion:
  //      xs[i][cin] = x[b][cin][k*120 + i - 2], i in [0,130)
  {
    const int cin = tid >> 1;
    const int part = tid & 1;
    const float* xp = x + ((size_t)b * CINC + cin) * TDIM;
    const int t0 = k * TK - 2;
#pragma unroll 4
    for (int jj = 0; jj < 17; ++jj) {
      const int i = part * 68 + jj * 4;
      if (i < 130) {
#pragma unroll
        for (int u = 0; u < 4; ++u) {
          const int ii = i + u;
          if (ii < 130) {
            const int gt = t0 + ii;
            float v = (gt >= 0 && gt < TDIM) ? xp[gt] : 0.0f;
            xs[ii * CINC + cin] = (__bf16)v;
          }
        }
      }
    }
  }
  async_wait0();
  __syncthreads();

  // wave tiling: wm in {0,1} -> 4 M tiles each; wn in {0..3} -> 2 N tiles each
  const int wm = wave & 1;
  const int wn = wave >> 1;
  const int nbase = wn * 32;
  v8f acc[4][2];
#pragma unroll
  for (int mt = 0; mt < 4; ++mt)
#pragma unroll
    for (int n = 0; n < 2; ++n) acc[mt][n] = v8f{};

#pragma unroll
  for (int s = 0; s < 12; ++s) {
    const int tap = s >> 2;                // unfold tap (causal delay)
    const int c32 = (s & 3) * 32;          // cin sub-block
    // B fragments: lane = cout column, K {0..7,16..23} / {8..15,24..31}
    v16bf bfrag[2];
#pragma unroll
    for (int n = 0; n < 2; ++n) {
      const __bf16* bp = wc + (s * 32 + lhalf * 8) * COUTC + nbase + n * 16 + l16;
#pragma unroll
      for (int i = 0; i < 8; ++i) {
        bfrag[n][i]     = bp[i * COUTC];
        bfrag[n][8 + i] = bp[(16 + i) * COUTC];
      }
    }
#pragma unroll
    for (int mt = 0; mt < 4; ++mt) {
      const int row = (wm * 4 + mt) * 16 + l16 + 2 - tap;  // xs time row
      const __bf16* apx = xs + row * CINC + c32 + lhalf * 8;
      v8bf alo = *(const v8bf*)apx;
      v8bf ahi = *(const v8bf*)(apx + 16);
      v16bf afrag = SHUF16(alo, ahi);
#pragma unroll
      for (int n = 0; n < 2; ++n)
        acc[mt][n] = __builtin_amdgcn_wmma_f32_16x16x32_bf16(
            false, afrag, false, bfrag[n], (short)0, acc[mt][n], false, false);
    }
  }

  float bv[2];
#pragma unroll
  for (int n = 0; n < 2; ++n)
    bv[n] = biasbuf[(size_t)chunk * COUTC + nbase + n * 16 + l16];

  // ---- stage D (+bias) into per-wave LDS region (reuses xs/wc after barrier)
  __syncthreads();
  float* stg = (float*)smem + wave * 2048;        // [64 t][32 cout] f32 = 8KB
#pragma unroll
  for (int mt = 0; mt < 4; ++mt) {
#pragma unroll
    for (int n = 0; n < 2; ++n) {
#pragma unroll
      for (int i = 0; i < 8; ++i) {
        const int lt = mt * 16 + i + 8 * lhalf;   // local t within wave strip
        stg[lt * 32 + n * 16 + l16] = acc[mt][n][i] + bv[n];
      }
    }
  }
  // ---- coalesced non-temporal writeback along t (wave strip = 64 t x 32 co)
  if (lane < 16 && (wm == 0 || lane < 14)) {      // global t = wm*64+lane*4 < 120
    const int lt = lane * 4;
    const int gt = wm * 64 + lt;
#pragma unroll
    for (int co = 0; co < 32; ++co) {
      v4f v;
#pragma unroll
      for (int u = 0; u < 4; ++u) v[u] = stg[(lt + u) * 32 + co];
      float* yp = y + ((size_t)(b * COUTC + nbase + co)) * TDIM + k * TK + gt;
      __builtin_nontemporal_store(v, (v4f*)yp);
    }
  }
}

// ---------------------------------------------------------------------------
extern "C" void kernel_launch(void* const* d_in, const int* in_sizes, int n_in,
                              void* d_out, int out_size, void* d_ws, size_t ws_size,
                              hipStream_t stream) {
  const float* x   = (const float*)d_in[0];
  const float* z   = (const float*)d_in[1];
  const float* w1  = (const float*)d_in[2];
  const float* b1  = (const float*)d_in[3];
  const float* w2  = (const float*)d_in[4];
  const float* b2  = (const float*)d_in[5];
  const float* bw1 = (const float*)d_in[6];
  const float* bb1 = (const float*)d_in[7];
  const float* bw2 = (const float*)d_in[8];
  const float* bb2 = (const float*)d_in[9];
  float* y = (float*)d_out;

  // Workspace layout (all 16B-aligned offsets):
  //   Hmat    bf16 [32][1600]      = 102,400 B
  //   biasbuf f32  [1600][128]     = 819,200 B
  //   Wall    bf16 [1600][49152]   = 157,286,400 B
  __bf16* Hmat    = (__bf16*)d_ws;
  float*  biasbuf = (float*)((char*)d_ws + 102400);
  __bf16* Wall    = (__bf16*)((char*)d_ws + 102400 + 819200);

  head_kernel<<<NCHUNK, 64, 0, stream>>>(z, w1, b1, bw1, bb1, bw2, bb2,
                                         Hmat, biasbuf);
  wgen_kernel<<<dim3(NCHUNK / 16, 16), 256, 0, stream>>>(w2, b2, Hmat, Wall);
  conv_kernel<<<NCHUNK, 256, 0, stream>>>(x, Wall, biasbuf, y);
}
